// Attention_83691732730020
// MI455X (gfx1250) — matrix operand order repaired and checked by reading, not compiled, not run on hardware
//
#include <hip/hip_runtime.h>

// ---------------------------------------------------------------------------
// MI455X (gfx1250) multi-stream cross-attention.
// - All matmuls via v_wmma_f32_16x16x32_bf16 (wave32 WMMA, f32 accum).
// - Attention K/V tiles staged to LDS by the Tensor Data Mover
//   (tensor_load_to_lds + s_wait_tensorcnt), with TDM LDS padding giving
//   bank-conflict-free strides.
// Workspace use: ~129 MB (bf16 weights, bf16 Q/K/V^T per stream, f32 ctx).
// ---------------------------------------------------------------------------

typedef unsigned short u16;
typedef unsigned int   u32;
typedef unsigned long long u64;

typedef __attribute__((ext_vector_type(16))) __bf16 v16bf;
typedef __attribute__((ext_vector_type(8)))  float  v8f;
typedef __attribute__((ext_vector_type(4)))  unsigned int v4u;
typedef __attribute__((ext_vector_type(8)))  int    v8i32;
typedef __attribute__((ext_vector_type(4)))  int    v4i32;

union FragB { v16bf v; u32 u[8]; };

__device__ inline v8f v8f_zero() {
    v8f z = {0.f,0.f,0.f,0.f,0.f,0.f,0.f,0.f};
    return z;
}

__device__ inline v8f wmma_bf16(const FragB& a, const FragB& b, v8f c) {
    return __builtin_amdgcn_wmma_f32_16x16x32_bf16(
        /*neg_a=*/false, a.v, /*neg_b=*/false, b.v,
        /*c_mod=*/(short)0, c, /*reuse_a=*/false, /*reuse_b=*/false);
}

__device__ inline u16 f32_to_bf16(float x) {            // RNE
    u32 a = __float_as_uint(x);
    a += 0x7fffu + ((a >> 16) & 1u);
    return (u16)(a >> 16);
}
__device__ inline u32 pack_bf16(float lo, float hi) {
    return (u32)f32_to_bf16(lo) | ((u32)f32_to_bf16(hi) << 16);
}

// Problem constants
#define B_SZ   16
#define D_SZ   768
#define H_SZ   12
#define DH_SZ  64
#define SCALE_ 0.125f

// LDS carve for attention (single __shared__ object => base offset 0)
#define K_STRIDE 144              // 32 keys x (128B row + 16B TDM pad)
#define V_STRIDE 80               // 64 dh   x ( 64B row + 16B TDM pad)
#define V_OFF    4608             // 32*144
#define P_OFF    9728             // 4608 + 64*80
#define SMEM_SZ  18176            // P_OFF + 4*16*33*4

// TDM group-1 flag dwords: data_size=2B (code 1), pad_enable,
// pad_interval code (2^(v+1) dwords), pad_amount code (v+1 dwords)
#define TDM_FLAGS_K ((1u<<16) | (1u<<20) | (4u<<22) | (3u<<25))  // 32 dw rows, +4 dw pad
#define TDM_FLAGS_V ((1u<<16) | (1u<<20) | (3u<<22) | (3u<<25))  // 16 dw rows, +4 dw pad

// 2D tensor tile -> LDS via Tensor Data Mover (D# per ISA 08 §8)
__device__ inline void tdm_load_2d(u32 lds_off, const void* gptr, u32 flags_d0,
                                   u32 tdim0, u32 tdim1, u32 tile0, u32 tile1,
                                   u32 stride0)
{
    u64 ga = (u64)(size_t)gptr;
    v4u g0 = { 1u,                                   // count=1, user mode
               lds_off,                              // D#.lds_addr
               (u32)ga,                              // global_addr[31:0]
               ((u32)(ga >> 32) & 0x01FFFFFFu) | (2u << 30) };  // [56:32] | type=2
    v8i32 g1 = {0,0,0,0,0,0,0,0};
    g1[0] = (int)flags_d0;                                        // mask=0, flags
    g1[1] = (int)((tdim0 & 0xffffu) << 16);                       // tensor_dim0 lo
    g1[2] = (int)(((tdim0 >> 16) & 0xffffu) | ((tdim1 & 0xffffu) << 16));
    g1[3] = (int)(((tdim1 >> 16) & 0xffffu) | ((tile0 & 0xffffu) << 16));
    g1[4] = (int)(tile1 & 0xffffu);                               // tile_dim2 = 0
    g1[5] = (int)stride0;                                         // dim0_stride lo
    v4i32 z4 = {0,0,0,0};
    v8i32 z8 = {0,0,0,0,0,0,0,0};
    // amdgpu-toolchain (clang-23) 6-arg form
    __builtin_amdgcn_tensor_load_to_lds(g0, g1, z4, z4, z8, 0);
}

// ---------------------------------------------------------------------------
// Kernel 0: fp32 -> bf16 elementwise (weight conversion)
// ---------------------------------------------------------------------------
__global__ void f32_to_bf16_kernel(const float* __restrict__ src,
                                   u16* __restrict__ dst, int n) {
    int i = blockIdx.x * blockDim.x + threadIdx.x;
    if (i < n) dst[i] = f32_to_bf16(src[i]);
}

// ---------------------------------------------------------------------------
// Kernel 1: QKV projection GEMM.  Y = X @ W + b, emitted as bf16 directly
// into attention-friendly layouts:
//   transposeV==0: Y[(b*H+h)*NP*64 + n*64 + dh]     (Q, K: row-major per head)
//   transposeV==1: Y[(b*H+h)*64*NP + dh*NP + n]     (V^T: keys contiguous)
// ---------------------------------------------------------------------------
__global__ __launch_bounds__(256)
void proj_qkv_kernel(const float* __restrict__ X, const u16* __restrict__ W,
                     const float* __restrict__ bias, u16* __restrict__ Y,
                     int N, int NP, int transposeV)
{
    __shared__ u16 Asub[32][40];   // [token][k]  stride 40 -> conflict-free
    __shared__ u16 Bsub[64][40];   // [ncol][k]   (W tile transposed)

    const int b    = blockIdx.z;
    const int h    = blockIdx.y;          // 64-col group == one head
    const int m0   = blockIdx.x * 32;     // token row base within batch b
    const int tid  = threadIdx.x;
    const int wave = tid >> 5, lane = tid & 31;
    const int mi   = wave & 1, ni = wave >> 1;   // 2x4 wave tiling
    const int m    = lane & 15, kh = lane >> 4;
    const int ncol0 = h * DH_SZ;

    v8f acc = v8f_zero();

    for (int k0 = 0; k0 < D_SZ; k0 += 32) {
        {   // A tile: 32 tokens x 32 k, f32 -> bf16
            int row = tid >> 3;
            int kk  = (tid & 7) * 4;
            int n   = m0 + row;
            float4 xv = make_float4(0.f, 0.f, 0.f, 0.f);
            if (n < N)
                xv = *(const float4*)(X + ((size_t)(b * N + n)) * D_SZ + k0 + kk);
            *(u32*)&Asub[row][kk]     = pack_bf16(xv.x, xv.y);
            *(u32*)&Asub[row][kk + 2] = pack_bf16(xv.z, xv.w);
        }
        {   // B tile: W[k0..k0+31][ncol0..ncol0+63] -> Bsub[n][k]
            int kk = tid & 31;
            int nb = (tid >> 5) * 8;
            #pragma unroll
            for (int j = 0; j < 4; ++j) {
                u32 w2 = *(const u32*)(W + ((size_t)(k0 + kk)) * D_SZ + ncol0 + nb + 2 * j);
                Bsub[nb + 2 * j][kk]     = (u16)(w2 & 0xffffu);
                Bsub[nb + 2 * j + 1][kk] = (u16)(w2 >> 16);
            }
        }
        __syncthreads();

        FragB a, bb;
        #pragma unroll
        for (int j = 0; j < 8; ++j) {
            int ka = kh * 16 + 2 * j;                                  // A layout
            a.u[j]  = *(const u32*)&Asub[mi * 16 + m][ka];
            int kb  = kh * 16 + 2 * j;                                 // B layout
            bb.u[j] = *(const u32*)&Bsub[ni * 16 + m][kb];
        }
        acc = wmma_bf16(a, bb, acc);
        __syncthreads();
    }

    const int dh = ni * 16 + m;
    const float bv = bias[ncol0 + dh];
    #pragma unroll
    for (int r = 0; r < 8; ++r) {
        int n = m0 + mi * 16 + kh * 8 + r;          // n < NP by construction
        float val = acc[r] + bv;
        size_t idx;
        if (!transposeV)
            idx = ((size_t)(b * H_SZ + h) * NP + n) * DH_SZ + dh;
        else
            idx = ((size_t)(b * H_SZ + h) * DH_SZ + dh) * (size_t)NP + n;
        Y[idx] = f32_to_bf16(val);
    }
}

// ---------------------------------------------------------------------------
// S-tile helper (global-memory path, used by the probs re-pass)
// ---------------------------------------------------------------------------
__device__ inline void qk_tiles(const FragB& aq0, const FragB& aq1,
                                const u16* __restrict__ K, int bh, int NPk,
                                int k0, int m, int kh, v8f& s0, v8f& s1)
{
    #pragma unroll
    for (int t = 0; t < 2; ++t) {
        const u16* kp = K + ((size_t)bh * NPk + k0 + t * 16 + m) * DH_SZ;
        FragB bk;
        v8f& st = t ? s1 : s0;
        #pragma unroll
        for (int j = 0; j < 8; ++j)
            bk.u[j] = *(const u32*)(kp + kh * 16 + 2 * j);            // dh 0..31
        st = wmma_bf16(aq0, bk, st);
        #pragma unroll
        for (int j = 0; j < 8; ++j)
            bk.u[j] = *(const u32*)(kp + 32 + kh * 16 + 2 * j);       // dh 32..63
        st = wmma_bf16(aq1, bk, st);
    }
}

// ---------------------------------------------------------------------------
// Kernel 2: fused 3-source attention for one query stream.
// ctx = sum_s softmax(Q K_s^T * scale) V_s   (independent softmax per source)
// One wave owns a 16-row query tile.  Per 32-key block, wave 0 stages the
// K tile (32x64 bf16) and V^T tile (64x32 bf16) to LDS with the TDM
// (padded rows for conflict-free WMMA fragment reads), then all 4 waves
// consume them.  Flash-style online softmax per source; img-self emits the
// globally-normalized probability tensor in a second S pass.
// ---------------------------------------------------------------------------
__global__ __launch_bounds__(128)
void attn_kernel(const u16* __restrict__ Q, int Nq, int NPq,
                 const u16* __restrict__ K0p, const u16* __restrict__ V0p, int Nk0, int NP0,
                 const u16* __restrict__ K1p, const u16* __restrict__ V1p, int Nk1, int NP1,
                 const u16* __restrict__ K2p, const u16* __restrict__ V2p, int Nk2, int NP2,
                 float* __restrict__ ctx, float* __restrict__ probs)
{
    __shared__ alignas(128) char smem[SMEM_SZ];   // only shared object -> LDS offset 0

    const int wave = threadIdx.x >> 5;
    const int lane = threadIdx.x & 31;
    const int bh   = blockIdx.y;
    const int m    = lane & 15;
    const int kh   = lane >> 4;

    // clamp OOB query tiles instead of returning: all waves join barriers;
    // duplicated waves write identical values to identical addresses (benign)
    int qt = blockIdx.x * 4 + wave;
    const int qtmax = (NPq >> 4) - 1;
    if (qt > qtmax) qt = qtmax;

    float* plds = (float*)(smem + P_OFF) + wave * (16 * 33);

    // Q fragments (A layout), two dh-chunks of 32
    FragB aq0, aq1;
    {
        const u16* qp = Q + ((size_t)(bh * NPq + qt * 16 + m)) * DH_SZ;
        #pragma unroll
        for (int j = 0; j < 8; ++j) {
            int k = kh * 16 + 2 * j;
            aq0.u[j] = *(const u32*)(qp + k);
            aq1.u[j] = *(const u32*)(qp + 32 + k);
        }
    }

    const u16* Ks[3]  = {K0p, K1p, K2p};
    const u16* Vs[3]  = {V0p, V1p, V2p};
    const int  Nks[3] = {Nk0, Nk1, Nk2};
    const int  NPs[3] = {NP0, NP1, NP2};

    v8f oacc[4];
    #pragma unroll
    for (int nt = 0; nt < 4; ++nt) oacc[nt] = v8f_zero();

    for (int s = 0; s < 3; ++s) {
        const u16* Kp  = Ks[s];
        const u16* VTp = Vs[s];
        const int  Nk  = Nks[s];
        const int  NPk = NPs[s];
        const int  nkb = NPk >> 5;

        float mrow[8], lrow[8];
        v8f o[4];
        #pragma unroll
        for (int r = 0; r < 8; ++r) { mrow[r] = -3.0e38f; lrow[r] = 0.f; }
        #pragma unroll
        for (int nt = 0; nt < 4; ++nt) o[nt] = v8f_zero();

        for (int kb = 0; kb < nkb; ++kb) {
            const int k0 = kb * 32;

            __syncthreads();                       // previous block's LDS reads done
            if (wave == 0) {
                // K tile: 32 rows (keys) x 64 elem, row stride 64 elem
                tdm_load_2d(0u, Kp + ((size_t)bh * NPk + k0) * DH_SZ,
                            TDM_FLAGS_K, 64u, 32u, 64u, 32u, 64u);
                // V^T tile: 64 rows (dh) x 32 elem (keys), row stride NPk elem
                tdm_load_2d((u32)V_OFF, VTp + (size_t)bh * DH_SZ * NPk + k0,
                            TDM_FLAGS_V, 32u, 64u, 32u, 64u, (u32)NPk);
                __builtin_amdgcn_s_wait_tensorcnt(0);
            }
            __syncthreads();                       // TDM data visible to all waves

            // S = Q K^T from LDS K tile
            v8f s0 = v8f_zero(), s1 = v8f_zero();
            #pragma unroll
            for (int t = 0; t < 2; ++t) {
                const char* kr = smem + (t * 16 + m) * K_STRIDE;
                FragB bk;
                v8f& st = t ? s1 : s0;
                #pragma unroll
                for (int j = 0; j < 8; ++j)
                    bk.u[j] = *(const u32*)(kr + (kh * 16 + 2 * j) * 2);
                st = wmma_bf16(aq0, bk, st);
                #pragma unroll
                for (int j = 0; j < 8; ++j)
                    bk.u[j] = *(const u32*)(kr + (32 + kh * 16 + 2 * j) * 2);
                st = wmma_bf16(aq1, bk, st);
            }

            // scale, key-mask, online softmax update
            const bool v0 = (k0 + m)      < Nk;
            const bool v1 = (k0 + 16 + m) < Nk;
            #pragma unroll
            for (int r = 0; r < 8; ++r) {
                float a  = v0 ? s0[r] * SCALE_ : -3.0e38f;
                float b2 = v1 ? s1[r] * SCALE_ : -3.0e38f;
                float mx = fmaxf(a, b2);
                #pragma unroll
                for (int off = 1; off < 16; off <<= 1)
                    mx = fmaxf(mx, __shfl_xor(mx, off, 32));
                float nm   = fmaxf(mrow[r], mx);
                float corr = __expf(mrow[r] - nm);
                float p0   = __expf(a - nm);
                float p1   = __expf(b2 - nm);
                float ps   = p0 + p1;
                #pragma unroll
                for (int off = 1; off < 16; off <<= 1)
                    ps += __shfl_xor(ps, off, 32);
                lrow[r] = lrow[r] * corr + ps;
                mrow[r] = nm;
                s0[r] = p0; s1[r] = p1;
                #pragma unroll
                for (int nt = 0; nt < 4; ++nt) o[nt][r] *= corr;
            }

            // stage P (C layout) to per-wave LDS, re-read as A fragment (bf16)
            #pragma unroll
            for (int r = 0; r < 8; ++r) {
                plds[(kh * 8 + r) * 33 + m]      = s0[r];
                plds[(kh * 8 + r) * 33 + 16 + m] = s1[r];
            }
            asm volatile("s_wait_dscnt 0" ::: "memory");
            FragB ap;
            #pragma unroll
            for (int j = 0; j < 8; ++j) {
                int k = kh * 16 + 2 * j;
                ap.u[j] = pack_bf16(plds[m * 33 + k], plds[m * 33 + k + 1]);
            }

            // O += P * V from LDS V^T tile
            #pragma unroll
            for (int nt = 0; nt < 4; ++nt) {
                const char* vr = smem + V_OFF + (nt * 16 + m) * V_STRIDE;
                FragB bv;
                #pragma unroll
                for (int j = 0; j < 8; ++j)
                    bv.u[j] = *(const u32*)(vr + (kh * 16 + 2 * j) * 2);
                o[nt] = wmma_bf16(ap, bv, o[nt]);
            }
        }

        float inv[8];
        #pragma unroll
        for (int r = 0; r < 8; ++r) inv[r] = 1.0f / lrow[r];
        #pragma unroll
        for (int nt = 0; nt < 4; ++nt)
            #pragma unroll
            for (int r = 0; r < 8; ++r)
                oacc[nt][r] += o[nt][r] * inv[r];

        // second pass: emit normalized probs for source 0 (img self-attn)
        if (s == 0 && probs != nullptr) {
            for (int kb = 0; kb < nkb; ++kb) {
                const int k0 = kb * 32;
                v8f s0 = v8f_zero(), s1 = v8f_zero();
                qk_tiles(aq0, aq1, Kp, bh, NPk, k0, m, kh, s0, s1);
                #pragma unroll
                for (int r = 0; r < 8; ++r) {
                    int row = qt * 16 + kh * 8 + r;
                    if (row < Nq) {
                        int key0 = k0 + m, key1 = k0 + 16 + m;
                        size_t base = ((size_t)bh * Nq + row) * (size_t)Nk;
                        if (key0 < Nk)
                            probs[base + key0] = __expf(s0[r] * SCALE_ - mrow[r]) * inv[r];
                        if (key1 < Nk)
                            probs[base + key1] = __expf(s1[r] * SCALE_ - mrow[r]) * inv[r];
                    }
                }
            }
        }
    }

    // write ctx (single store; later GEMM guards padded rows)
    #pragma unroll
    for (int nt = 0; nt < 4; ++nt)
        #pragma unroll
        for (int r = 0; r < 8; ++r) {
            int row = qt * 16 + kh * 8 + r;
            ctx[((size_t)bh * NPq + row) * DH_SZ + nt * 16 + m] = oacc[nt][r];
        }
}

// ---------------------------------------------------------------------------
// Kernel 3: output projection.  out = merge(ctx) @ Wo + bo  (f32 output)
// ctx gather: token (b,n), col k -> ctx[(b*H + k/64)*NP*64 + n*64 + k%64]
// ---------------------------------------------------------------------------
__global__ __launch_bounds__(256)
void proj_out_kernel(const float* __restrict__ CTX, const u16* __restrict__ W,
                     const float* __restrict__ bias, float* __restrict__ Y,
                     int N, int NP)
{
    __shared__ u16 Asub[32][40];
    __shared__ u16 Bsub[64][40];

    const int b    = blockIdx.z;
    const int h    = blockIdx.y;
    const int m0   = blockIdx.x * 32;
    const int tid  = threadIdx.x;
    const int wave = tid >> 5, lane = tid & 31;
    const int mi   = wave & 1, ni = wave >> 1;
    const int m    = lane & 15, kh = lane >> 4;
    const int ncol0 = h * DH_SZ;

    v8f acc = v8f_zero();

    for (int k0 = 0; k0 < D_SZ; k0 += 32) {
        {   // A tile from ctx (k0 is 32-aligned -> single head per tile)
            int row = tid >> 3;
            int kk  = (tid & 7) * 4;
            int n   = m0 + row;
            float4 xv = make_float4(0.f, 0.f, 0.f, 0.f);
            if (n < N) {
                const float* src = CTX +
                    (((size_t)(b * H_SZ + (k0 >> 6)) * NP + n) * DH_SZ + (k0 & 63) + kk);
                xv = *(const float4*)src;
            }
            *(u32*)&Asub[row][kk]     = pack_bf16(xv.x, xv.y);
            *(u32*)&Asub[row][kk + 2] = pack_bf16(xv.z, xv.w);
        }
        {
            int kk = tid & 31;
            int nb = (tid >> 5) * 8;
            #pragma unroll
            for (int j = 0; j < 4; ++j) {
                u32 w2 = *(const u32*)(W + ((size_t)(k0 + kk)) * D_SZ + ncol0 + nb + 2 * j);
                Bsub[nb + 2 * j][kk]     = (u16)(w2 & 0xffffu);
                Bsub[nb + 2 * j + 1][kk] = (u16)(w2 >> 16);
            }
        }
        __syncthreads();

        FragB a, bb;
        #pragma unroll
        for (int j = 0; j < 8; ++j) {
            int ka = kh * 16 + 2 * j;
            a.u[j]  = *(const u32*)&Asub[mi * 16 + m][ka];
            int kb  = kh * 16 + 2 * j;
            bb.u[j] = *(const u32*)&Bsub[ni * 16 + m][kb];
        }
        acc = wmma_bf16(a, bb, acc);
        __syncthreads();
    }

    const int dh = ni * 16 + m;
    const float bv = bias[ncol0 + dh];
    #pragma unroll
    for (int r = 0; r < 8; ++r) {
        int n = m0 + mi * 16 + kh * 8 + r;
        if (n < N)
            Y[((size_t)(b * N + n)) * D_SZ + ncol0 + dh] = acc[r] + bv;
    }
}

// ---------------------------------------------------------------------------
// Host orchestration
// ---------------------------------------------------------------------------
extern "C" void kernel_launch(void* const* d_in, const int* in_sizes, int n_in,
                              void* d_out, int out_size, void* d_ws, size_t ws_size,
                              hipStream_t stream)
{
    (void)in_sizes; (void)n_in; (void)out_size; (void)ws_size;

    constexpr int N_I = 577, N_T = 256, N_C = 64;
    constexpr int NP_I = 608, NP_T = 256, NP_C = 64;           // padded to 32
    constexpr size_t WSZ  = (size_t)D_SZ * D_SZ;               // 589824
    constexpr size_t NE_I = (size_t)B_SZ * H_SZ * NP_I * DH_SZ;
    constexpr size_t NE_T = (size_t)B_SZ * H_SZ * NP_T * DH_SZ;
    constexpr size_t NE_C = (size_t)B_SZ * H_SZ * NP_C * DH_SZ;
    constexpr size_t O_TXT = (size_t)B_SZ * N_I * D_SZ;
    constexpr size_t O_CLN = O_TXT + (size_t)B_SZ * N_T * D_SZ;
    constexpr size_t O_W   = O_CLN + (size_t)B_SZ * N_C * D_SZ;

    const float* Ximg = (const float*)d_in[0];
    const float* Xtxt = (const float*)d_in[1];
    const float* Xcln = (const float*)d_in[2];
    float* out = (float*)d_out;

    // workspace carve (aligned to 256 B)
    char* ws = (char*)d_ws;
    size_t off = 0;
    auto carve = [&](size_t bytes) -> char* {
        char* p = ws + off;
        off += (bytes + 255) & ~(size_t)255;
        return p;
    };
    u16*   wbf   = (u16*)carve(12 * WSZ * sizeof(u16));
    u16*   Qi    = (u16*)carve(NE_I * sizeof(u16));
    u16*   Ki    = (u16*)carve(NE_I * sizeof(u16));
    u16*   VTi   = (u16*)carve(NE_I * sizeof(u16));
    u16*   Qt    = (u16*)carve(NE_T * sizeof(u16));
    u16*   Kt    = (u16*)carve(NE_T * sizeof(u16));
    u16*   VTt   = (u16*)carve(NE_T * sizeof(u16));
    u16*   Qc    = (u16*)carve(NE_C * sizeof(u16));
    u16*   Kc    = (u16*)carve(NE_C * sizeof(u16));
    u16*   VTc   = (u16*)carve(NE_C * sizeof(u16));
    float* ctx_i = (float*)carve(NE_I * sizeof(float));
    float* ctx_t = (float*)carve(NE_T * sizeof(float));
    float* ctx_c = (float*)carve(NE_C * sizeof(float));

    // 1) weights -> bf16 (L2-resident: 14 MB of 192 MB)
    for (int i = 0; i < 12; ++i) {
        const float* wsrc = (const float*)d_in[3 + 2 * i];
        f32_to_bf16_kernel<<<dim3((unsigned)(WSZ / 256)), dim3(256), 0, stream>>>(
            wsrc, wbf + (size_t)i * WSZ, (int)WSZ);
    }

    // 2) QKV projections (9 GEMMs).  weight idx: 0..3 img q/k/v/o, 4..7 txt, 8..11 cln
    struct PJ { const float* X; int widx; int bidx; u16* out; int N, NP, tv; };
    const PJ pj[9] = {
        {Ximg, 0,  4,  Qi,  N_I, NP_I, 0}, {Ximg, 1,  6,  Ki,  N_I, NP_I, 0}, {Ximg, 2,  8,  VTi, N_I, NP_I, 1},
        {Xtxt, 4, 12,  Qt,  N_T, NP_T, 0}, {Xtxt, 5, 14,  Kt,  N_T, NP_T, 0}, {Xtxt, 6, 16,  VTt, N_T, NP_T, 1},
        {Xcln, 8, 20,  Qc,  N_C, NP_C, 0}, {Xcln, 9, 22,  Kc,  N_C, NP_C, 0}, {Xcln, 10, 24, VTc, N_C, NP_C, 1},
    };
    for (int i = 0; i < 9; ++i) {
        dim3 grid(pj[i].NP / 32, H_SZ, B_SZ);
        proj_qkv_kernel<<<grid, dim3(256), 0, stream>>>(
            pj[i].X, wbf + (size_t)pj[i].widx * WSZ,
            (const float*)d_in[pj[i].bidx], pj[i].out, pj[i].N, pj[i].NP, pj[i].tv);
    }

    // 3) fused attention: one launch per query stream (3 KV sources each)
    {
        dim3 grid((NP_I / 16 + 3) / 4, B_SZ * H_SZ);
        attn_kernel<<<grid, dim3(128), 0, stream>>>(
            Qi, N_I, NP_I,
            Ki, VTi, N_I, NP_I,   // source 0 = self -> probs emitted
            Kt, VTt, N_T, NP_T,
            Kc, VTc, N_C, NP_C,
            ctx_i, out + O_W);
    }
    {
        dim3 grid((NP_T / 16 + 3) / 4, B_SZ * H_SZ);
        attn_kernel<<<grid, dim3(128), 0, stream>>>(
            Qt, N_T, NP_T,
            Kt, VTt, N_T, NP_T,
            Ki, VTi, N_I, NP_I,
            Kc, VTc, N_C, NP_C,
            ctx_t, nullptr);
    }
    {
        dim3 grid((NP_C / 16 + 3) / 4, B_SZ * H_SZ);
        attn_kernel<<<grid, dim3(128), 0, stream>>>(
            Qc, N_C, NP_C,
            Kc, VTc, N_C, NP_C,
            Ki, VTi, N_I, NP_I,
            Kt, VTt, N_T, NP_T,
            ctx_c, nullptr);
    }

    // 4) output projections
    proj_out_kernel<<<dim3(NP_I / 32, H_SZ, B_SZ), dim3(256), 0, stream>>>(
        ctx_i, wbf + 3 * WSZ, (const float*)d_in[10], out, N_I, NP_I);
    proj_out_kernel<<<dim3(NP_T / 32, H_SZ, B_SZ), dim3(256), 0, stream>>>(
        ctx_t, wbf + 7 * WSZ, (const float*)d_in[18], out + O_TXT, N_T, NP_T);
    proj_out_kernel<<<dim3(NP_C / 32, H_SZ, B_SZ), dim3(256), 0, stream>>>(
        ctx_c, wbf + 11 * WSZ, (const float*)d_in[26], out + O_CLN, N_C, NP_C);
}
